// MultiHeadSelfAttention_2705829396630
// MI455X (gfx1250) — compile-verified, matrix operand order checked by reading, not run on hardware
//
#include <hip/hip_runtime.h>
#include <hip/hip_bf16.h>

// ---------------------------------------------------------------------------
// MHA forward for MI455X (gfx1250, wave32, WMMA 16x16x32 f16 -> f32)
//   1) cvt f32->f16 : hidden_states, Wq, Wk, Wv, Wo
//   2) WMMA GEMM    : Q/K/V = X @ W^T   (TDM tensor_load_to_lds, double buffer)
//   3) RoPE + head transpose -> [b,h,t,d] f16
//   4) Flash attention (online softmax, WMMA for QK^T and PV)
//   5) WMMA GEMM    : out = ctx @ Wo^T  (f32 output)
// ---------------------------------------------------------------------------

#if __has_include(<hip/amd_detail/amd_gfx1250_TDM.h>)
#define TDM_ARGS6 1   // therock/clang-23 headers -> 6-arg tensor builtin
#endif

typedef _Float16 half_t;
typedef __attribute__((ext_vector_type(16))) _Float16 v16h;
typedef __attribute__((ext_vector_type(8)))  _Float16 v8h;
typedef __attribute__((ext_vector_type(8)))  float    v8f;
typedef __attribute__((ext_vector_type(4)))  unsigned int v4u;
typedef __attribute__((ext_vector_type(8)))  int      v8i;
typedef __attribute__((ext_vector_type(4)))  int      v4i;

#define HIDDEN   2048
#define NHEADS   16
#define HD       128
#define BB       2
#define TT       2048

// D = A(16x32 f16) * B(32x16 f16) + C(16x16 f32)
__device__ __forceinline__ v8f wmma32(v16h a, v16h b, v8f c) {
    return __builtin_amdgcn_wmma_f32_16x16x32_f16(
        /*neg_a=*/false, a, /*neg_b=*/false, b,
        /*c_mod=*/(short)0, c, /*reuse_a=*/false, /*reuse_b=*/false);
}

// Load one A/B fragment (16x32 / 32x16, f16). rp = pointer to this lane's
// row (fixed M for A, fixed N for B), k-contiguous. Per ISA layout:
// halves 0..7  -> K = o..o+7 ; halves 8..15 -> K = o+16..o+23 ;
// o = kbase + (lane>=16 ? 8 : 0)
__device__ __forceinline__ v16h frag_ld(const half_t* rp, int lh, int kbase) {
    v16h f;
    const int o = kbase + lh * 8;
    ((v8h*)&f)[0] = *(const v8h*)(rp + o);
    ((v8h*)&f)[1] = *(const v8h*)(rp + o + 16);
    return f;
}

// reductions across the 16 lanes of a half-wave (rows live within a half)
__device__ __forceinline__ float rowmax16(float v) {
    v = fmaxf(v, __shfl_xor(v, 1));
    v = fmaxf(v, __shfl_xor(v, 2));
    v = fmaxf(v, __shfl_xor(v, 4));
    v = fmaxf(v, __shfl_xor(v, 8));
    return v;
}
__device__ __forceinline__ float rowsum16(float v) {
    v += __shfl_xor(v, 1);
    v += __shfl_xor(v, 2);
    v += __shfl_xor(v, 4);
    v += __shfl_xor(v, 8);
    return v;
}

// ---------------------------------------------------------------------------
// TDM: DMA a 2-D tile (tile_rows x tile_d0_u32 dwords, 4-byte units) from
// global row-major memory into contiguous LDS at lds_off.  D# per ISA ch.8.
// All inputs are workgroup-uniform -> descriptor lands in SGPRs.
__device__ __forceinline__ void tdm_load_tile_2d(
    unsigned int lds_off, const half_t* gptr,
    unsigned int tile_d0_u32, unsigned int tile_rows,
    unsigned int row_stride_u32) {
    const unsigned long long ga = (unsigned long long)(uintptr_t)gptr;

    v4u g0;
    g0.x = 1u;                                    // count=1, is_restore=0
    g0.y = lds_off;                               // LDS byte address
    g0.z = (unsigned int)ga;                      // global_addr[31:0]
    g0.w = (unsigned int)((ga >> 32) & 0x01FFFFFFu) | 0x80000000u; // [56:32]|type=2

    const unsigned long long ds  = 2;             // data_size = 4 bytes
    const unsigned long long td0 = 0x100000;      // tensor dims: generous,
    const unsigned long long td1 = 0x100000;      // tiles are always in-bounds
    const unsigned long long str = row_stride_u32;

    // group1 bit packing (256 bits as 4x u64):
    //  [17:16] data_size  [79:48] tensor_dim0  [111:80] tensor_dim1
    //  [127:112] tile_dim0 [143:128] tile_dim1 [207:160] tensor_dim0_stride
    const unsigned long long w0 = (ds << 16) | ((td0 & 0xFFFFull) << 48);
    const unsigned long long w1 = (td0 >> 16) | (td1 << 16) |
                                  ((unsigned long long)tile_d0_u32 << 48);
    const unsigned long long w2 = (unsigned long long)tile_rows |
                                  ((str & 0xFFFFFFFFull) << 32);
    const unsigned long long w3 = (str >> 32) & 0xFFFFull;   // dim1_stride = 0

    v8i g1;
    g1[0] = (int)(unsigned int)w0;  g1[1] = (int)(unsigned int)(w0 >> 32);
    g1[2] = (int)(unsigned int)w1;  g1[3] = (int)(unsigned int)(w1 >> 32);
    g1[4] = (int)(unsigned int)w2;  g1[5] = (int)(unsigned int)(w2 >> 32);
    g1[6] = (int)(unsigned int)w3;  g1[7] = (int)(unsigned int)(w3 >> 32);

    const v4i gz = {0, 0, 0, 0};
#ifdef TDM_ARGS6
    const v8i gz8 = {0, 0, 0, 0, 0, 0, 0, 0};
    __builtin_amdgcn_tensor_load_to_lds(g0, g1, gz, gz, gz8, 0);
#else
    __builtin_amdgcn_tensor_load_to_lds(g0, g1, gz, gz, 0);
#endif
}

// ---------------------------------------------------------------------------
__global__ __launch_bounds__(256)
void cvt_f32_f16(const float* __restrict__ in, half_t* __restrict__ out,
                 long long n) {
    long long i = (long long)blockIdx.x * blockDim.x + threadIdx.x;
    long long stride = (long long)gridDim.x * blockDim.x;
    for (; i < n; i += stride) out[i] = (half_t)in[i];
}

// ---------------------------------------------------------------------------
// C[M,N] = A[M,K] * Bw[N,K]^T ; 128x128 tile / block (8 waves), K-step 32.
// Tiles arrive via TDM (tensor_load_to_lds), double buffered; wave 0 issues
// and drains TENSORcnt, workgroup barrier publishes the buffer.
template <bool OUT_F32>
__global__ __launch_bounds__(256)
void gemm_wmma(const half_t* __restrict__ A, const half_t* __restrict__ Bw,
               void* __restrict__ Cp, int M, int N, int K) {
    __shared__ __align__(16) half_t sA[2][128 * 32];   // [m][k] 8 KB x2
    __shared__ __align__(16) half_t sB[2][128 * 32];   // [n][k] 8 KB x2

    const int tid = threadIdx.x;
    const int wv = tid >> 5, l = tid & 31, lh = l >> 4, ln = l & 15;
    const int mg = wv & 3;            // 4 wave rows * 32
    const int ng = wv >> 2;           // 2 wave cols * 64
    const int mBase = blockIdx.y * 128, nBase = blockIdx.x * 128;

    const unsigned int ldsA = (unsigned int)(uintptr_t)&sA[0][0];
    const unsigned int ldsB = (unsigned int)(uintptr_t)&sB[0][0];
    const unsigned int strideU = (unsigned int)(K / 2);   // row stride, dwords

    v8f acc[2][4] = {};

    if (wv == 0) {
        tdm_load_tile_2d(ldsA, A  + (size_t)mBase * K, 16, 128, strideU);
        tdm_load_tile_2d(ldsB, Bw + (size_t)nBase * K, 16, 128, strideU);
        __builtin_amdgcn_s_wait_tensorcnt(0);
    }
    __syncthreads();

    int cur = 0;
    for (int kk = 0; kk < K; kk += 32) {
        const bool more = (kk + 32) < K;
        if (more && wv == 0) {   // async-prefetch next K tile into other buffer
            tdm_load_tile_2d(ldsA + (cur ^ 1) * 8192,
                             A  + (size_t)mBase * K + kk + 32, 16, 128, strideU);
            tdm_load_tile_2d(ldsB + (cur ^ 1) * 8192,
                             Bw + (size_t)nBase * K + kk + 32, 16, 128, strideU);
        }

        v16h af[2], bf[4];
        #pragma unroll
        for (int mt = 0; mt < 2; ++mt)
            af[mt] = frag_ld(&sA[cur][(mg * 32 + mt * 16 + ln) * 32], lh, 0);
        #pragma unroll
        for (int nt = 0; nt < 4; ++nt)
            bf[nt] = frag_ld(&sB[cur][(ng * 64 + nt * 16 + ln) * 32], lh, 0);
        #pragma unroll
        for (int mt = 0; mt < 2; ++mt)
            #pragma unroll
            for (int nt = 0; nt < 4; ++nt)
                acc[mt][nt] = wmma32(af[mt], bf[nt], acc[mt][nt]);

        __syncthreads();                       // everyone done reading 'cur'
        if (more && wv == 0)
            __builtin_amdgcn_s_wait_tensorcnt(0);
        __syncthreads();                       // next buffer visible to all
        cur ^= 1;
    }

    #pragma unroll
    for (int mt = 0; mt < 2; ++mt)
        #pragma unroll
        for (int nt = 0; nt < 4; ++nt)
            #pragma unroll
            for (int r = 0; r < 8; ++r) {
                const int row = mBase + mg * 32 + mt * 16 + r + lh * 8;
                const int col = nBase + ng * 64 + nt * 16 + ln;
                if constexpr (OUT_F32)
                    ((float*)Cp)[(size_t)row * N + col] = acc[mt][nt][r];
                else
                    ((half_t*)Cp)[(size_t)row * N + col] = (half_t)acc[mt][nt][r];
            }
}

// ---------------------------------------------------------------------------
// RoPE on Q,K + transpose Q,K,V from [b,t,h,d] to [b,h,t,d] (f16).
__global__ __launch_bounds__(256)
void rope_transpose(const half_t* __restrict__ Q, const half_t* __restrict__ Kk,
                    const half_t* __restrict__ V,
                    const float* __restrict__ cosc, const float* __restrict__ sinc,
                    const long long* __restrict__ pos_ids,
                    half_t* __restrict__ QT, half_t* __restrict__ KT,
                    half_t* __restrict__ VT) {
    const long long idx = (long long)blockIdx.x * blockDim.x + threadIdx.x;
    const long long total = (long long)BB * TT * NHEADS * (HD / 2);
    if (idx >= total) return;
    const int i2 = (int)(idx & 63);
    const int h  = (int)((idx >> 6) & 15);
    const int t  = (int)((idx >> 10) % TT);
    const int b  = (int)((idx >> 10) / TT);
    const int pos = (int)pos_ids[t];
    const int d0 = 2 * i2, d1 = d0 + 1;

    const size_t src = ((size_t)(b * TT + t)) * HIDDEN + (size_t)h * HD;
    const size_t dst = (((size_t)(b * NHEADS + h)) * TT + t) * HD;

    const float c0 = cosc[(size_t)pos * HD + d0], c1 = cosc[(size_t)pos * HD + d1];
    const float s0 = sinc[(size_t)pos * HD + d0], s1 = sinc[(size_t)pos * HD + d1];

    const float q0 = (float)Q[src + d0], q1 = (float)Q[src + d1];
    QT[dst + d0] = (half_t)(q0 * c0 - q1 * s0);
    QT[dst + d1] = (half_t)(q1 * c1 + q0 * s1);

    const float k0 = (float)Kk[src + d0], k1 = (float)Kk[src + d1];
    KT[dst + d0] = (half_t)(k0 * c0 - k1 * s0);
    KT[dst + d1] = (half_t)(k1 * c1 + k0 * s1);

    VT[dst + d0] = V[src + d0];
    VT[dst + d1] = V[src + d1];
}

// ---------------------------------------------------------------------------
// Flash attention: block = (b,h,q-block of 64 rows), 128 threads = 4 waves,
// each wave owns 16 q-rows; K-tile = 32 keys so P is exactly one A-fragment.
__global__ __launch_bounds__(128)
void attn_kernel(const half_t* __restrict__ QT, const half_t* __restrict__ KT,
                 const half_t* __restrict__ VT, half_t* __restrict__ ctx) {
    __shared__ __align__(16) half_t sK[32 * HD];       // [kpos][d]  8 KB
    __shared__ __align__(16) half_t sV[HD * 32];       // [d][kpos]  8 KB
    __shared__ __align__(16) half_t sP[4 * 16 * 32];   // per-wave P 4 KB

    const int qb = blockIdx.x;           // T/64
    const int h  = blockIdx.y;
    const int b  = blockIdx.z;
    const int tid = threadIdx.x;
    const int wv = tid >> 5, l = tid & 31, lh = l >> 4, ln = l & 15;

    const half_t* Qb = QT + ((size_t)(b * NHEADS + h)) * TT * HD;
    const half_t* Kb = KT + ((size_t)(b * NHEADS + h)) * TT * HD;
    const half_t* Vb = VT + ((size_t)(b * NHEADS + h)) * TT * HD;

    const int qrow = qb * 64 + wv * 16 + ln;   // this lane's A-matrix row

    v16h qf[4];
    #pragma unroll
    for (int c = 0; c < 4; ++c)
        qf[c] = frag_ld(Qb + (size_t)qrow * HD, lh, c * 32);

    v8f oacc[8] = {};
    float mrun[8], lrun[8];
    #pragma unroll
    for (int r = 0; r < 8; ++r) { mrun[r] = -__builtin_inff(); lrun[r] = 0.f; }

    const int nkt = qb * 2 + 2;                // causal: keys <= q-block end
    const float scale = 0.08838834764831845f;  // 1/sqrt(128)

    for (int kt = 0; kt < nkt; ++kt) {
        const int k0 = kt * 32;
        __syncthreads();
        #pragma unroll
        for (int i = 0; i < 4; ++i) {
            const int id = tid * 4 + i;        // 0..511
            const int row = id >> 4;           // key pos 0..31
            const int seg = id & 15;           // d segment
            *(v8h*)(sK + row * HD + seg * 8) =
                *(const v8h*)(Kb + (size_t)(k0 + row) * HD + seg * 8);
            v8h vv = *(const v8h*)(Vb + (size_t)(k0 + row) * HD + seg * 8);
            #pragma unroll
            for (int j = 0; j < 8; ++j)
                sV[(seg * 8 + j) * 32 + row] = vv[j];
        }
        __syncthreads();

        // S = Q * K^T  (two 16-key column tiles)
        v8f sacc[2] = {};
        #pragma unroll
        for (int j = 0; j < 2; ++j)
            #pragma unroll
            for (int c = 0; c < 4; ++c) {
                v16h kf = frag_ld(sK + (j * 16 + ln) * HD, lh, c * 32);
                sacc[j] = wmma32(qf[c], kf, sacc[j]);
            }

        // online softmax update (rows of the C tile live in half-waves)
        float p0[8], p1[8], alpha[8];
        #pragma unroll
        for (int r = 0; r < 8; ++r) {
            const int row_q = qb * 64 + wv * 16 + r + lh * 8;
            const int c0 = k0 + ln, c1 = k0 + 16 + ln;
            float s0 = sacc[0][r] * scale; if (c0 > row_q) s0 = -1.0e9f;
            float s1 = sacc[1][r] * scale; if (c1 > row_q) s1 = -1.0e9f;
            float mx = rowmax16(fmaxf(s0, s1));
            const float mnew = fmaxf(mrun[r], mx);
            const float a = __expf(mrun[r] - mnew);
            p0[r] = __expf(s0 - mnew);
            p1[r] = __expf(s1 - mnew);
            const float rs = rowsum16(p0[r] + p1[r]);
            lrun[r] = lrun[r] * a + rs;
            mrun[r] = mnew;
            alpha[r] = a;
        }
        #pragma unroll
        for (int t = 0; t < 8; ++t)
            #pragma unroll
            for (int r = 0; r < 8; ++r)
                oacc[t][r] *= alpha[r];

        // P (C-layout) -> LDS -> A-fragment layout
        half_t* pw = sP + wv * (16 * 32);
        #pragma unroll
        for (int r = 0; r < 8; ++r) {
            pw[(r + lh * 8) * 32 + ln]      = (half_t)p0[r];
            pw[(r + lh * 8) * 32 + 16 + ln] = (half_t)p1[r];
        }
        // cross-lane LDS dependency inside one wave: explicit DS-counter wait
        asm volatile("s_wait_dscnt 0" ::: "memory");
        v16h pf = frag_ld(pw + ln * 32, lh, 0);

        #pragma unroll
        for (int t = 0; t < 8; ++t) {
            v16h vf = frag_ld(sV + (t * 16 + ln) * 32, lh, 0);
            oacc[t] = wmma32(pf, vf, oacc[t]);
        }
    }

    // normalize + store ctx as [b, t, h, d]  (matches transpose+reshape)
    #pragma unroll
    for (int t = 0; t < 8; ++t)
        #pragma unroll
        for (int r = 0; r < 8; ++r) {
            const int row_q = qb * 64 + wv * 16 + r + lh * 8;
            const int d = t * 16 + ln;
            const float v = oacc[t][r] / lrun[r];
            ctx[((size_t)(b * TT + row_q)) * HIDDEN + (size_t)h * HD + d] = (half_t)v;
        }
}

// ---------------------------------------------------------------------------
extern "C" void kernel_launch(void* const* d_in, const int* in_sizes, int n_in,
                              void* d_out, int out_size, void* d_ws, size_t ws_size,
                              hipStream_t stream) {
    const float*     hs   = (const float*)d_in[0];
    /* d_in[1] = attention_mask (pure causal; computed analytically) */
    const long long* pids = (const long long*)d_in[2];
    const float*     Wq   = (const float*)d_in[3];
    const float*     Wk   = (const float*)d_in[4];
    const float*     Wv   = (const float*)d_in[5];
    const float*     Wo   = (const float*)d_in[6];
    const float*     cosc = (const float*)d_in[7];
    const float*     sinc = (const float*)d_in[8];

    const long long NX = (long long)BB * TT * HIDDEN;          // 8.4M
    const long long NW = (long long)HIDDEN * HIDDEN;           // 4.2M
    const size_t MB = 1024ull * 1024ull;

    char* ws = (char*)d_ws;
    half_t* Xh  = (half_t*)(ws + 0 * MB);     // 16 MB
    half_t* Wqh = (half_t*)(ws + 16 * MB);    //  8 MB
    half_t* Wkh = (half_t*)(ws + 24 * MB);
    half_t* Wvh = (half_t*)(ws + 32 * MB);
    half_t* Woh = (half_t*)(ws + 40 * MB);
    half_t* Qh  = (half_t*)(ws + 48 * MB);    // 16 MB  (later reused for ctx)
    half_t* Kh  = (half_t*)(ws + 64 * MB);
    half_t* Vh  = (half_t*)(ws + 80 * MB);
    half_t* QTp = (half_t*)(ws + 96 * MB);
    half_t* KTp = (half_t*)(ws + 112 * MB);
    half_t* VTp = (half_t*)(ws + 128 * MB);

    // 1) f32 -> f16 conversions
    cvt_f32_f16<<<4096, 256, 0, stream>>>(hs, Xh, NX);
    cvt_f32_f16<<<4096, 256, 0, stream>>>(Wq, Wqh, NW);
    cvt_f32_f16<<<4096, 256, 0, stream>>>(Wk, Wkh, NW);
    cvt_f32_f16<<<4096, 256, 0, stream>>>(Wv, Wvh, NW);
    cvt_f32_f16<<<4096, 256, 0, stream>>>(Wo, Woh, NW);

    // 2) Q/K/V projections: [4096,2048] x [2048,2048]^T
    const int M = BB * TT, N = HIDDEN, K = HIDDEN;
    dim3 ggrid(N / 128, M / 128);
    gemm_wmma<false><<<ggrid, 256, 0, stream>>>(Xh, Wqh, (void*)Qh, M, N, K);
    gemm_wmma<false><<<ggrid, 256, 0, stream>>>(Xh, Wkh, (void*)Kh, M, N, K);
    gemm_wmma<false><<<ggrid, 256, 0, stream>>>(Xh, Wvh, (void*)Vh, M, N, K);

    // 3) RoPE + head transpose
    {
        const long long total = (long long)BB * TT * NHEADS * (HD / 2);
        const int blocks = (int)((total + 255) / 256);
        rope_transpose<<<blocks, 256, 0, stream>>>(Qh, Kh, Vh, cosc, sinc, pids,
                                                   QTp, KTp, VTp);
    }

    // 4) attention -> ctx [b,t,h,d] f16 (reuse Qh region; Q/K/V no longer read)
    half_t* ctx = Qh;
    dim3 agrid(TT / 64, NHEADS, BB);
    attn_kernel<<<agrid, 128, 0, stream>>>(QTp, KTp, VTp, ctx);

    // 5) output projection -> f32 d_out
    gemm_wmma<true><<<ggrid, 256, 0, stream>>>(ctx, Woh, d_out, M, N, K);

    (void)in_sizes; (void)n_in; (void)out_size; (void)ws_size;
}
